// DataNODE_84026740179392
// MI455X (gfx1250) — matrix operand order repaired and checked by reading, not compiled, not run on hardware
//
#include <hip/hip_runtime.h>
#include <hip/hip_bf16.h>

typedef _Float16 half_t;
typedef __attribute__((ext_vector_type(16))) _Float16 v16h;
typedef __attribute__((ext_vector_type(8)))  _Float16 v8h;
typedef __attribute__((ext_vector_type(2)))  __fp16   v2fp16;   // return type of cvt_pkrtz
typedef __attribute__((ext_vector_type(8)))  float    v8f;

#define HID 128
#define WAVES 4           // 128 threads/block, 16 points per wave -> 64 points/block

// ---------------- fast activations on native HW transcendentals ----------------
#define LOG2E 1.44269504088896f

__device__ __forceinline__ float fast_sigmoid(float x) {
    // 1/(1+exp(-x)) : v_exp_f32 + v_add + v_rcp_f32
    float e = __builtin_amdgcn_exp2f(x * -LOG2E);
    return __builtin_amdgcn_rcpf(1.0f + e);
}

__device__ __forceinline__ float fast_tanh(float x) {
    // tanh(x) = 1 - 2/(exp(2x)+1)
    float e = __builtin_amdgcn_exp2f(x * (2.0f * LOG2E));
    return 1.0f - 2.0f * __builtin_amdgcn_rcpf(e + 1.0f);
}

// ---------------- LDS fragment loaders (documented CDNA5 WMMA layouts) ----------------

// A-matrix 16x32 f16 (M x K), lane m=lane&15:
//   lanes 0-15 : K = 0..7 (V0-3), K = 16..23 (V4-7)
//   lanes16-31 : K = 8..15,       K = 24..31
__device__ __forceinline__ v16h lds_load_A(const half_t* W, int ldk, int mt, int kc, int lane) {
    int m  = mt * 16 + (lane & 15);
    int hi = (lane >> 4) & 1;
    const half_t* p = W + m * ldk + kc * 32 + hi * 8;
    v8h lo  = *(const v8h*)(p);
    v8h hiv = *(const v8h*)(p + 16);
    v16h a;
#pragma unroll
    for (int i = 0; i < 8; ++i) { a[i] = lo[i]; a[8 + i] = hiv[i]; }
    return a;
}

// B-matrix 32x16 f16 (K x N), lane n=lane&15 holds one column:
//   lanes 0-15 : K = 0..7, 16..23 ; lanes 16-31 : K = 8..15, 24..31  (matches A)
__device__ __forceinline__ v16h lds_load_B(const half_t* Z, int ldk, int kc, int lane) {
    int n  = lane & 15;
    int hi = (lane >> 4) & 1;
    const half_t* p = Z + n * ldk + kc * 32 + hi * 8;
    v8h lo  = *(const v8h*)(p);
    v8h hiv = *(const v8h*)(p + 16);
    v16h b;
#pragma unroll
    for (int i = 0; i < 8; ++i) { b[i] = lo[i]; b[8 + i] = hiv[i]; }
    return b;
}

// C 16x16 f32: VGPR r, lanes 0-15 -> M=r, N=lane ; lanes 16-31 -> M=8+r, N=lane-16

// ---------------- gated layer: Zout = relu( (Wh Z + bh) * sigmoid(Wg Z + bg) ) ----------------
template <int KC, int LDK>
__device__ __forceinline__ void gated_layer(const half_t* Wh, const half_t* Wg,
                                            const float* bh, const float* bg,
                                            const half_t* Zin, half_t* Zout, int lane) {
    v16h bfrag[KC];
#pragma unroll
    for (int kc = 0; kc < KC; ++kc) bfrag[kc] = lds_load_B(Zin, LDK, kc, lane);

    int n = lane & 15;
    int mo = ((lane >> 4) & 1) * 8;
#pragma unroll
    for (int mt = 0; mt < 8; ++mt) {
        v8f ch = {}; v8f cg = {};
#pragma unroll
        for (int kc = 0; kc < KC; ++kc) {
            v16h ah = lds_load_A(Wh, LDK, mt, kc, lane);
            ch = __builtin_amdgcn_wmma_f32_16x16x32_f16(false, ah, false, bfrag[kc],
                                                        (short)0, ch, false, false);
            v16h ag = lds_load_A(Wg, LDK, mt, kc, lane);
            cg = __builtin_amdgcn_wmma_f32_16x16x32_f16(false, ag, false, bfrag[kc],
                                                        (short)0, cg, false, false);
        }
        int mbase = mt * 16 + mo;
#pragma unroll
        for (int r = 0; r < 8; r += 2) {
            float h0 = ch[r]     + bh[mbase + r];
            float h1 = ch[r + 1] + bh[mbase + r + 1];
            float g0 = cg[r]     + bg[mbase + r];
            float g1 = cg[r + 1] + bg[mbase + r + 1];
            float z0 = fmaxf(h0 * fast_sigmoid(g0), 0.0f);
            float z1 = fmaxf(h1 * fast_sigmoid(g1), 0.0f);
            v2fp16 pk = __builtin_amdgcn_cvt_pkrtz(z0, z1);    // v_cvt_pk_rtz_f16_f32
            *(v2fp16*)(&Zout[n * HID + mbase + r]) = pk;       // ds_store_b32
        }
    }
}

struct LdsCtx {
    const half_t *WH1, *WG1, *WH2, *WG2, *WOUT;
    const float  *bh1, *bg1, *bh2, *bg2, *bout;
    half_t *zin, *zh;
};

// one evaluation of the velocity field: (y, t) -> tanh(out)*0.5
__device__ __forceinline__ void field_eval(const LdsCtx& c, int lane, float t,
                                           float y0, float y1, float y2,
                                           float& k0, float& k1, float& k2) {
    if (lane < 16) {
        union { v2fp16 v[2]; unsigned long long u; } pk;
        pk.v[0] = __builtin_amdgcn_cvt_pkrtz(y0, y1);
        pk.v[1] = __builtin_amdgcn_cvt_pkrtz(y2, t);
        *(unsigned long long*)(&c.zin[lane * 32]) = pk.u;      // ds_store_b64
    }
    gated_layer<1, 32 >(c.WH1, c.WG1, c.bh1, c.bg1, c.zin, c.zh, lane);
    gated_layer<4, HID>(c.WH2, c.WG2, c.bh2, c.bg2, c.zh,  c.zh, lane);   // in-place: B preloaded

    v8f acc = {};
#pragma unroll
    for (int kc = 0; kc < 4; ++kc) {
        v16h b = lds_load_B(c.zh, HID, kc, lane);
        v16h a = lds_load_A(c.WOUT, HID, 0, kc, lane);
        acc = __builtin_amdgcn_wmma_f32_16x16x32_f16(false, a, false, b,
                                                     (short)0, acc, false, false);
    }
    int mb = ((lane >> 4) & 1) * 8;   // lanes 16-31 compute padded rows (unused)
    k0 = fast_tanh(acc[0] + c.bout[mb + 0]) * 0.5f;
    k1 = fast_tanh(acc[1] + c.bout[mb + 1]) * 0.5f;
    k2 = fast_tanh(acc[2] + c.bout[mb + 2]) * 0.5f;
}

// LDS plan (bytes):
//  WH1 8192 | WG1 8192 | WH2 32768 | WG2 32768 | WOUT 4096 | biases 2112
//  ZIN 4*16*32*2 = 4096 | ZH 4*16*128*2 = 16384   -> total 108608
#define SMEM_BYTES 108608

__global__ __launch_bounds__(128) void node_rk4_wmma_kernel(
    const float* __restrict__ x0,
    const float* __restrict__ h1_w, const float* __restrict__ h1_b,
    const float* __restrict__ g1_w, const float* __restrict__ g1_b,
    const float* __restrict__ h2_w, const float* __restrict__ h2_b,
    const float* __restrict__ g2_w, const float* __restrict__ g2_b,
    const float* __restrict__ out_w, const float* __restrict__ out_b,
    float* __restrict__ out, int Npts, int P)
{
    extern __shared__ __attribute__((aligned(32))) char smem[];
    half_t* WH1  = (half_t*)smem;                  // [128][32] (K padded 4->32)
    half_t* WG1  = WH1 + 128 * 32;
    half_t* WH2  = WG1 + 128 * 32;                 // [128][128]
    half_t* WG2  = WH2 + 128 * 128;
    half_t* WOUT = WG2 + 128 * 128;                // [16][128] (M padded 3->16)
    float*  bh1  = (float*)(WOUT + 16 * 128);
    float*  bg1  = bh1 + 128;
    float*  bh2  = bg1 + 128;
    float*  bg2  = bh2 + 128;
    float*  bout = bg2 + 128;                      // [16]
    half_t* ZIN  = (half_t*)(bout + 16);           // per-wave [16][32]
    half_t* ZH   = ZIN + WAVES * 16 * 32;          // per-wave [16][128]

    const int tid = threadIdx.x;

    // ---- stage weights to LDS in f16 (done once, reused by 4 RK evals) ----
    for (int i = tid; i < 128 * 128; i += 128) {
        WH2[i] = (half_t)h2_w[i];
        WG2[i] = (half_t)g2_w[i];
    }
    for (int i = tid; i < 128 * 32; i += 128) {
        int m = i >> 5, k = i & 31;
        WH1[i] = (half_t)((k < 4) ? h1_w[m * 4 + k] : 0.0f);
        WG1[i] = (half_t)((k < 4) ? g1_w[m * 4 + k] : 0.0f);
    }
    for (int i = tid; i < 16 * 128; i += 128) {
        int m = i >> 7, k = i & 127;
        WOUT[i] = (half_t)((m < 3) ? out_w[m * 128 + k] : 0.0f);
    }
    { bh1[tid] = h1_b[tid]; bg1[tid] = g1_b[tid]; bh2[tid] = h2_b[tid]; bg2[tid] = g2_b[tid]; }
    if (tid < 16) bout[tid] = (tid < 3) ? out_b[tid] : 0.0f;
    for (int i = tid; i < WAVES * 16 * 32; i += 128) ZIN[i] = (half_t)0.0f;  // pad channels stay 0
    __syncthreads();

    const int wave = tid >> 5, lane = tid & 31;
    LdsCtx c;
    c.WH1 = WH1; c.WG1 = WG1; c.WH2 = WH2; c.WG2 = WG2; c.WOUT = WOUT;
    c.bh1 = bh1; c.bg1 = bg1; c.bh2 = bh2; c.bg2 = bg2; c.bout = bout;
    c.zin = ZIN + wave * 16 * 32;
    c.zh  = ZH  + wave * 16 * 128;

    // ---- load this lane's point (lanes 16-31 mirror lanes 0-15; they only feed WMMA) ----
    const int n  = lane & 15;
    const int p  = blockIdx.x * (WAVES * 16) + wave * 16 + n;
    const int pc = (p < P) ? p : (P - 1);
    const int bb = pc / Npts;
    const int nn = pc - bb * Npts;
    const size_t base = (size_t)bb * 3 * Npts + nn;
    const float xa = x0[base];
    const float xb = x0[base + Npts];
    const float xc = x0[base + 2 * (size_t)Npts];

    // ---- RK4 (3/8 rule), h = 1 ----
    const float th = 1.0f / 3.0f;
    float k10,k11,k12, k20,k21,k22, k30,k31,k32, k40,k41,k42;

    field_eval(c, lane, 0.0f, xa, xb, xc, k10, k11, k12);
    field_eval(c, lane, th,
               xa + th * k10, xb + th * k11, xc + th * k12, k20, k21, k22);
    field_eval(c, lane, 2.0f * th,
               xa + (k20 - th * k10), xb + (k21 - th * k11), xc + (k22 - th * k12),
               k30, k31, k32);
    field_eval(c, lane, 1.0f,
               xa + (k10 - k20 + k30), xb + (k11 - k21 + k31), xc + (k12 - k22 + k32),
               k40, k41, k42);

    if (lane < 16 && p < P) {
        out[base]                    = xa + (k10 + 3.0f * (k20 + k30) + k40) * 0.125f;
        out[base + Npts]             = xb + (k11 + 3.0f * (k21 + k31) + k41) * 0.125f;
        out[base + 2 * (size_t)Npts] = xc + (k12 + 3.0f * (k22 + k32) + k42) * 0.125f;
    }
}

extern "C" void kernel_launch(void* const* d_in, const int* in_sizes, int n_in,
                              void* d_out, int out_size, void* d_ws, size_t ws_size,
                              hipStream_t stream) {
    const float* x0   = (const float*)d_in[0];
    const float* h1_w = (const float*)d_in[1];
    const float* h1_b = (const float*)d_in[2];
    const float* g1_w = (const float*)d_in[3];
    const float* g1_b = (const float*)d_in[4];
    const float* h2_w = (const float*)d_in[5];
    const float* h2_b = (const float*)d_in[6];
    const float* g2_w = (const float*)d_in[7];
    const float* g2_b = (const float*)d_in[8];
    const float* ow   = (const float*)d_in[9];
    const float* ob   = (const float*)d_in[10];
    float* out = (float*)d_out;

    const int Npts = 50000;                 // reference: x0 is [B, 3, N] with N = 50000
    const int P    = in_sizes[0] / 3;       // B * N total points
    const int blocks = (P + WAVES * 16 - 1) / (WAVES * 16);

    node_rk4_wmma_kernel<<<blocks, 128, SMEM_BYTES, stream>>>(
        x0, h1_w, h1_b, g1_w, g1_b, h2_w, h2_b, g2_w, g2_b, ow, ob,
        out, Npts, P);
}
